// Prot3DGraphModel_2224793060106
// MI455X (gfx1250) — matrix operand order repaired and checked by reading, not compiled, not run on hardware
//
#include <hip/hip_runtime.h>
#include <stdint.h>

// ---------------------------------------------------------------------------
// Prot3DGraphModel forward for MI455X (gfx1250, wave32).
//   - All dense linears run as bf16 WMMA GEMMs (v_wmma_f32_16x16x32_bf16),
//     f32 accumulate, weights/activations converted to bf16 once per call.
//   - proj_edge is algebraically folded into each layer's edge linear:
//       e_l = edge_s @ (We_l @ Wpe)^T + (We_l @ bpe + be_l)
//     eliminating the [E,128] "ea" tensor and shrinking K from 128 to 39.
//   - Edge softmax: order-preserving u32 key + atomicMax for segment max,
//     atomicAdd for denominators and message scatter.
// Input order assumption (setup_inputs dict insertion order, params nested
// insertion order): seq, node_s, seq_emb, edge_index, edge_s, batch,
// embed, pn.W, pn.b, pe.W, pe.b, then per layer q.W q.b k.W k.b v.W v.b
// e.W e.b skip.W skip.b  (n_in = 41).
// ---------------------------------------------------------------------------

#define DEVFN __device__ __forceinline__

typedef __attribute__((ext_vector_type(16))) __bf16 v16bf;
typedef __attribute__((ext_vector_type(8)))  float  v8f;

union FragU { uint4 u4[2]; v16bf v; };

static constexpr int N_NODES = 40000;
static constexpr int N_EDGES = 600000;
static constexpr int N_GRAPH = 128;
static constexpr int K0_PAD  = 1312;   // 20+6+1280 = 1306 padded to 32-multiple
static constexpr int KE_PAD  = 64;     // 39 padded

DEVFN uint16_t f2bf(float f) {
  uint32_t u = __float_as_uint(f);
  uint32_t r = u + 0x7FFFu + ((u >> 16) & 1u);   // round-to-nearest-even
  return (uint16_t)(r >> 16);
}

// ---- weight prep ----------------------------------------------------------

__global__ void pack_weight_bf16(const float* __restrict__ W,
                                 uint16_t* __restrict__ dst,
                                 int rows, int K, int Kpad) {
  int t = blockIdx.x * 256 + threadIdx.x;
  if (t >= rows * Kpad) return;
  int r = t / Kpad, kk = t - r * Kpad;
  dst[t] = (kk < K) ? f2bf(W[(size_t)r * K + kk]) : (uint16_t)0;
}

// Wcomb = We @ Wpe  (cout x 39, padded to 64), bcomb = We @ bpe + be
__global__ void fold_edge_weight(const float* __restrict__ We,
                                 const float* __restrict__ Wpe,
                                 const float* __restrict__ bpe,
                                 const float* __restrict__ be,
                                 uint16_t* __restrict__ Wcomb,
                                 float* __restrict__ bcomb, int cout) {
  int n = blockIdx.x, kk = threadIdx.x;          // grid=cout, block=64
  if (n >= cout) return;
  float s = 0.f;
  if (kk < 39)
    for (int j = 0; j < 128; ++j) s += We[(size_t)n * 128 + j] * Wpe[j * 39 + kk];
  Wcomb[(size_t)n * KE_PAD + kk] = (kk < 39) ? f2bf(s) : (uint16_t)0;
  if (kk == 0) {
    float sb = be[n];
    for (int j = 0; j < 128; ++j) sb += We[(size_t)n * 128 + j] * bpe[j];
    bcomb[n] = sb;
  }
}

// ---- activation staging ---------------------------------------------------

__global__ void build_x0(const int* __restrict__ seq,
                         const float* __restrict__ node_s,
                         const float* __restrict__ seq_emb,
                         const float* __restrict__ embed,
                         uint16_t* __restrict__ X0) {
  int row = blockIdx.x;
  int sidx = seq[row];
  uint16_t* out = X0 + (size_t)row * K0_PAD;
  for (int col = threadIdx.x; col < K0_PAD; col += 256) {
    float v;
    if (col < 20)        v = embed[sidx * 20 + col];
    else if (col < 26)   v = node_s[(size_t)row * 6 + (col - 20)];
    else if (col < 1306) v = seq_emb[(size_t)row * 1280 + (col - 26)];
    else                 v = 0.f;
    out[col] = f2bf(v);
  }
}

__global__ void build_xe(const float* __restrict__ edge_s,
                         uint16_t* __restrict__ Xe) {
  int row = blockIdx.x, col = threadIdx.x;       // grid=E, block=64
  float v = (col < 39) ? edge_s[(size_t)row * 39 + col] : 0.f;
  Xe[(size_t)row * KE_PAD + col] = f2bf(v);
}

// ---- bf16 WMMA GEMM: C[M,Nc] = A[M,K](bf16) @ B[Nc,K](bf16)^T + bias ------
// One wave computes one 16x16 tile. Fragment loads are contiguous per lane
// per the 16-bit A (16x32) / B (32x16) VGPR layouts of v_wmma_f32_16x16x32.

__global__ void __launch_bounds__(256)
gemm_bf16_wmma(const uint16_t* __restrict__ A, int lda,
               const uint16_t* __restrict__ B, int K,
               const float* __restrict__ bias,
               float* __restrict__ Cf, uint16_t* __restrict__ Cbf,
               int M, int Nc) {
  const int lane   = threadIdx.x & 31;
  const int wave   = threadIdx.x >> 5;
  const int tilesN = Nc >> 4;
  const int tile   = blockIdx.x * 8 + wave;
  if (tile >= (M >> 4) * tilesN) return;         // whole wave exits together
  const int m0 = (tile / tilesN) << 4;
  const int n0 = (tile % tilesN) << 4;
  const int r = lane & 15, h = lane >> 4;

  const uint16_t* arow = A + (size_t)(m0 + r) * lda;
  const uint16_t* brow = B + (size_t)(n0 + r) * K;

  v8f acc = {};
  for (int k0 = 0; k0 < K; k0 += 32) {
    FragU a, b;
    a.u4[0] = *(const uint4*)(arow + k0 + h * 8);        // K 0-7 / 8-15
    a.u4[1] = *(const uint4*)(arow + k0 + 16 + h * 8);   // K 16-23 / 24-31
    b.u4[0] = *(const uint4*)(brow + k0 + h * 8);        // K 0-7 / 8-15
    b.u4[1] = *(const uint4*)(brow + k0 + 16 + h * 8);   // K 16-23 / 24-31
    acc = __builtin_amdgcn_wmma_f32_16x16x32_bf16(
        false, a.v, false, b.v, (short)0, acc, false, false);
  }

  const float bv = bias ? bias[n0 + r] : 0.f;            // bias is per-column
#pragma unroll
  for (int i = 0; i < 8; ++i) {
    float val = acc[i] + bv;
    size_t off = (size_t)(m0 + i + 8 * h) * Nc + (n0 + r);
    if (Cf)  Cf[off]  = val;
    if (Cbf) Cbf[off] = f2bf(val);
  }
}

// ---- edge attention (wave32 per edge) -------------------------------------

DEVFN uint32_t fkey(float f) {                   // order-preserving f32->u32
  uint32_t u = __float_as_uint(f);
  return (u & 0x80000000u) ? ~u : (u | 0x80000000u);
}
DEVFN float fkey_inv(uint32_t k) {
  return (k & 0x80000000u) ? __uint_as_float(k ^ 0x80000000u)
                           : __uint_as_float(~k);
}

__global__ void __launch_bounds__(256)
edge_alpha(const int* __restrict__ src, const int* __restrict__ dst,
           const float* __restrict__ q, const float* __restrict__ k,
           const float* __restrict__ e, float* __restrict__ alpha,
           unsigned* __restrict__ mkey, int C, float scale) {
  int edge = blockIdx.x * 8 + (threadIdx.x >> 5);
  if (edge >= N_EDGES) return;
  int lane = threadIdx.x & 31;
  int s = src[edge], d = dst[edge];
  const float* qd = q + (size_t)d * C;
  const float* ks = k + (size_t)s * C;
  const float* ee = e + (size_t)edge * C;
  float sum = 0.f;
  for (int c = lane; c < C; c += 32) sum += qd[c] * (ks[c] + ee[c]);
  for (int off = 16; off > 0; off >>= 1) sum += __shfl_xor(sum, off, 32);
  if (lane == 0) {
    float al = sum * scale;
    alpha[edge] = al;
    atomicMax(mkey + d, fkey(al));
  }
}

__global__ void edge_exp(const int* __restrict__ dst,
                         const float* __restrict__ alpha,
                         const unsigned* __restrict__ mkey,
                         float* __restrict__ a, float* __restrict__ denom) {
  int e = blockIdx.x * 256 + threadIdx.x;
  if (e >= N_EDGES) return;
  int d = dst[e];
  float av = expf(alpha[e] - fkey_inv(mkey[d]));
  a[e] = av;
  atomicAdd(denom + d, av);
}

__global__ void __launch_bounds__(256)
edge_msg(const int* __restrict__ src, const int* __restrict__ dst,
         const float* __restrict__ v, const float* __restrict__ e,
         const float* __restrict__ a, const float* __restrict__ denom,
         float* __restrict__ acc, int C) {
  int edge = blockIdx.x * 8 + (threadIdx.x >> 5);
  if (edge >= N_EDGES) return;
  int lane = threadIdx.x & 31;
  int s = src[edge], d = dst[edge];
  float w = a[edge] / denom[d];
  const float* vs = v + (size_t)s * C;
  const float* ee = e + (size_t)edge * C;
  float* ad = acc + (size_t)d * C;
  for (int c = lane; c < C; c += 32) atomicAdd(ad + c, w * (vs[c] + ee[c]));
}

// ---- pointwise + pooling --------------------------------------------------

__global__ void leaky_write(const float* __restrict__ acc,
                            float* __restrict__ xf,
                            uint16_t* __restrict__ xbf, int C) {
  int row = blockIdx.x;
  for (int c = threadIdx.x; c < C; c += 256) {
    float v = acc[(size_t)row * C + c];
    v = v > 0.f ? v : 0.01f * v;
    xf [(size_t)row * C + c] = v;
    xbf[(size_t)row * C + c] = f2bf(v);
  }
}

__global__ void pool_sum(const float* __restrict__ xf,
                         const int* __restrict__ batch,
                         float* __restrict__ gsum, float* __restrict__ gcnt) {
  int row = blockIdx.x;
  int g = batch[row];
  for (int c = threadIdx.x; c < 256; c += 256)
    atomicAdd(gsum + (size_t)g * 256 + c, xf[(size_t)row * 256 + c]);
  if (threadIdx.x == 0) atomicAdd(gcnt + g, 1.0f);
}

__global__ void pool_div(const float* __restrict__ gsum,
                         const float* __restrict__ gcnt,
                         float* __restrict__ out) {
  int t = blockIdx.x * 256 + threadIdx.x;
  if (t >= N_GRAPH * 256) return;
  out[t] = gsum[t] / fmaxf(gcnt[t >> 8], 1.0f);
}

// ---------------------------------------------------------------------------

extern "C" void kernel_launch(void* const* d_in, const int* in_sizes, int n_in,
                              void* d_out, int out_size, void* d_ws, size_t ws_size,
                              hipStream_t stream) {
  (void)in_sizes; (void)n_in; (void)out_size; (void)ws_size;
  const int*   seq     = (const int*)d_in[0];
  const float* node_s  = (const float*)d_in[1];
  const float* seq_emb = (const float*)d_in[2];
  const int*   eidx    = (const int*)d_in[3];
  const float* edge_s  = (const float*)d_in[4];
  const int*   batch   = (const int*)d_in[5];
  const float* embed   = (const float*)d_in[6];
  const float* Wpn     = (const float*)d_in[7];
  const float* bpn     = (const float*)d_in[8];
  const float* Wpe     = (const float*)d_in[9];
  const float* bpe     = (const float*)d_in[10];
  const float *Wq[3], *bq[3], *Wk[3], *bk[3], *Wv[3], *bv[3],
              *We[3], *be[3], *Ws[3], *bs[3];
  for (int l = 0; l < 3; ++l) {
    int base = 11 + l * 10;
    Wq[l] = (const float*)d_in[base + 0]; bq[l] = (const float*)d_in[base + 1];
    Wk[l] = (const float*)d_in[base + 2]; bk[l] = (const float*)d_in[base + 3];
    Wv[l] = (const float*)d_in[base + 4]; bv[l] = (const float*)d_in[base + 5];
    We[l] = (const float*)d_in[base + 6]; be[l] = (const float*)d_in[base + 7];
    Ws[l] = (const float*)d_in[base + 8]; bs[l] = (const float*)d_in[base + 9];
  }
  const int* src = eidx;
  const int* dst = eidx + N_EDGES;
  const int CIN[3]  = {128, 128, 256};
  const int COUT[3] = {128, 256, 256};

  // workspace bump allocator (256B aligned; total ~1.05 GB)
  uint8_t* p = (uint8_t*)d_ws;
  auto alloc = [&](size_t bytes) -> void* {
    void* r = p; p += (bytes + 255) & ~(size_t)255; return r;
  };
  uint16_t* wpn_bf = (uint16_t*)alloc((size_t)128 * K0_PAD * 2);
  uint16_t* wcomb[3]; float* bcomb[3];
  uint16_t *wq_bf[3], *wk_bf[3], *wv_bf[3], *wsk_bf[3];
  for (int l = 0; l < 3; ++l) {
    wcomb[l]  = (uint16_t*)alloc((size_t)COUT[l] * KE_PAD * 2);
    bcomb[l]  = (float*)   alloc((size_t)COUT[l] * 4);
    wq_bf[l]  = (uint16_t*)alloc((size_t)COUT[l] * CIN[l] * 2);
    wk_bf[l]  = (uint16_t*)alloc((size_t)COUT[l] * CIN[l] * 2);
    wv_bf[l]  = (uint16_t*)alloc((size_t)COUT[l] * CIN[l] * 2);
    wsk_bf[l] = (uint16_t*)alloc((size_t)COUT[l] * CIN[l] * 2);
  }
  uint16_t* X0   = (uint16_t*)alloc((size_t)N_NODES * K0_PAD * 2);
  uint16_t* Xe   = (uint16_t*)alloc((size_t)N_EDGES * KE_PAD * 2);
  uint16_t* xbf  = (uint16_t*)alloc((size_t)N_NODES * 256 * 2);
  float*    xf   = (float*)   alloc((size_t)N_NODES * 256 * 4);
  float*    qf   = (float*)   alloc((size_t)N_NODES * 256 * 4);
  float*    kf   = (float*)   alloc((size_t)N_NODES * 256 * 4);
  float*    vf   = (float*)   alloc((size_t)N_NODES * 256 * 4);
  float*    ef   = (float*)   alloc((size_t)N_EDGES * 256 * 4);
  float*    accf = (float*)   alloc((size_t)N_NODES * 256 * 4);
  float*    alpha= (float*)   alloc((size_t)N_EDGES * 4);
  float*    aexp = (float*)   alloc((size_t)N_EDGES * 4);
  unsigned* mkey = (unsigned*)alloc((size_t)N_NODES * 4);
  float*    denom= (float*)   alloc((size_t)N_NODES * 4);
  float*    gsum = (float*)   alloc((size_t)N_GRAPH * 256 * 4);
  float*    gcnt = (float*)   alloc((size_t)N_GRAPH * 4);

  auto packW = [&](const float* W, uint16_t* dstp, int rows, int K, int Kpad) {
    int total = rows * Kpad;
    pack_weight_bf16<<<(total + 255) / 256, 256, 0, stream>>>(W, dstp, rows, K, Kpad);
  };
  auto gemm = [&](const uint16_t* A, int lda, const uint16_t* B, int K,
                  const float* bias, float* Cf, uint16_t* Cbf, int M, int Nc) {
    int tiles = (M / 16) * (Nc / 16);
    gemm_bf16_wmma<<<(tiles + 7) / 8, 256, 0, stream>>>(A, lda, B, K, bias, Cf, Cbf, M, Nc);
  };

  // weight prep
  packW(Wpn, wpn_bf, 128, 1306, K0_PAD);
  for (int l = 0; l < 3; ++l) {
    packW(Wq[l], wq_bf[l],  COUT[l], CIN[l], CIN[l]);
    packW(Wk[l], wk_bf[l],  COUT[l], CIN[l], CIN[l]);
    packW(Wv[l], wv_bf[l],  COUT[l], CIN[l], CIN[l]);
    packW(Ws[l], wsk_bf[l], COUT[l], CIN[l], CIN[l]);
    fold_edge_weight<<<COUT[l], 64, 0, stream>>>(We[l], Wpe, bpe, be[l],
                                                 wcomb[l], bcomb[l], COUT[l]);
  }

  // activation staging + node projection
  build_x0<<<N_NODES, 256, 0, stream>>>(seq, node_s, seq_emb, embed, X0);
  build_xe<<<N_EDGES, 64, 0, stream>>>(edge_s, Xe);
  gemm(X0, K0_PAD, wpn_bf, K0_PAD, bpn, nullptr, xbf, N_NODES, 128);

  // 3 TransformerConv layers
  for (int l = 0; l < 3; ++l) {
    int cin = CIN[l], cout = COUT[l];
    gemm(xbf, cin, wq_bf[l],  cin,    bq[l],    qf,   nullptr, N_NODES, cout);
    gemm(xbf, cin, wk_bf[l],  cin,    bk[l],    kf,   nullptr, N_NODES, cout);
    gemm(xbf, cin, wv_bf[l],  cin,    bv[l],    vf,   nullptr, N_NODES, cout);
    gemm(Xe,  KE_PAD, wcomb[l], KE_PAD, bcomb[l], ef,  nullptr, N_EDGES, cout);
    gemm(xbf, cin, wsk_bf[l], cin,    bs[l],    accf, nullptr, N_NODES, cout); // skip into acc

    hipMemsetAsync(mkey,  0, (size_t)N_NODES * 4, stream);
    hipMemsetAsync(denom, 0, (size_t)N_NODES * 4, stream);
    float scale = 1.0f / sqrtf((float)cout);
    edge_alpha<<<(N_EDGES + 7) / 8, 256, 0, stream>>>(src, dst, qf, kf, ef,
                                                      alpha, mkey, cout, scale);
    edge_exp<<<(N_EDGES + 255) / 256, 256, 0, stream>>>(dst, alpha, mkey, aexp, denom);
    edge_msg<<<(N_EDGES + 7) / 8, 256, 0, stream>>>(src, dst, vf, ef, aexp, denom,
                                                    accf, cout);
    leaky_write<<<N_NODES, 256, 0, stream>>>(accf, xf, xbf, cout);
  }

  // global mean pool
  hipMemsetAsync(gsum, 0, (size_t)N_GRAPH * 256 * 4, stream);
  hipMemsetAsync(gcnt, 0, (size_t)N_GRAPH * 4, stream);
  pool_sum<<<N_NODES, 256, 0, stream>>>(xf, batch, gsum, gcnt);
  pool_div<<<N_GRAPH, 256, 0, stream>>>(gsum, gcnt, (float*)d_out);
}
